// SelfBlock_26474178412769
// MI455X (gfx1250) — matrix operand order repaired and checked by reading, not compiled, not run on hardware
//
#include <hip/hip_runtime.h>

typedef __bf16 bf16;
typedef __attribute__((ext_vector_type(16))) __bf16 v16bf;
typedef __attribute__((ext_vector_type(8)))  __bf16 bf16x8;
typedef __attribute__((ext_vector_type(8)))  float   v8f;
typedef __attribute__((ext_vector_type(4)))  int     v4i;

#define AS_GLOBAL __attribute__((address_space(1)))
#define AS_LDS    __attribute__((address_space(3)))

#if __has_builtin(__builtin_amdgcn_global_load_async_to_lds_b128)
  #define HAVE_ASYNC 1
#elif __has_builtin(__builtin_amdgcn_load_to_lds)
  #define HAVE_ASYNC 2
#else
  #define HAVE_ASYNC 0
#endif

// Copy 16 bytes global -> LDS. Async (ASYNCcnt-tracked DMA) when available.
static __device__ __forceinline__ void cp16_g2l(void* lds, const void* g) {
#if HAVE_ASYNC == 1
  __builtin_amdgcn_global_load_async_to_lds_b128(
      (AS_GLOBAL v4i*)(size_t)g,             // global b128 source
      (AS_LDS v4i*)(unsigned)(size_t)lds,    // ISA: LDS addr = flat addr[31:0]
      0, 0);
#elif HAVE_ASYNC == 2
  __builtin_amdgcn_load_to_lds((void*)g, (AS_LDS void*)(unsigned)(size_t)lds,
                               16, 0, 0);
#else
  *(bf16x8*)lds = *(const bf16x8*)g;
#endif
}

#if HAVE_ASYNC == 1
  #if __has_builtin(__builtin_amdgcn_s_wait_asynccnt)
    #define WAIT_ASYNC(N) __builtin_amdgcn_s_wait_asynccnt(N)
  #else
    #define WAIT_ASYNC(N) asm volatile("s_wait_asynccnt " #N ::: "memory")
  #endif
#else
  #define WAIT_ASYNC(N) ((void)0)   // sync fallback: dscnt handled by compiler
#endif

static __device__ __forceinline__ v16bf cat16(bf16x8 lo, bf16x8 hi) {
  return __builtin_shufflevector(lo, hi, 0,1,2,3,4,5,6,7,8,9,10,11,12,13,14,15);
}
static __device__ __forceinline__ bf16x8 ld8(const bf16* p) {
  return *(const bf16x8*)p;
}
static __device__ __forceinline__ v8f wmma_bf16(v16bf a, v16bf b, v8f c) {
  return __builtin_amdgcn_wmma_f32_16x16x32_bf16(false, a, false, b, (short)0, c, false, false);
}
static __device__ __forceinline__ v8f zero8() {
  v8f z = {0.f,0.f,0.f,0.f,0.f,0.f,0.f,0.f};
  return z;
}

// ---------------------------------------------------------------------------
// fp32 -> bf16 converters
// ---------------------------------------------------------------------------
__global__ __launch_bounds__(256)
void cvt_bf16(const float* __restrict__ in, bf16* __restrict__ out, long n) {
  long stride = (long)gridDim.x * blockDim.x;
  for (long i = (long)blockIdx.x * blockDim.x + threadIdx.x; i < n; i += stride)
    out[i] = (bf16)in[i];
}

__global__ __launch_bounds__(256)
void cvt_bf16_strided(const float* __restrict__ in, bf16* __restrict__ out,
                      long rows, long cols, long ldo) {
  long total = rows * cols;
  long stride = (long)gridDim.x * blockDim.x;
  for (long i = (long)blockIdx.x * blockDim.x + threadIdx.x; i < total; i += stride) {
    long r = i / cols, c = i - r * cols;
    out[r * ldo + c] = (bf16)in[i];
  }
}

// ---------------------------------------------------------------------------
// WMMA GEMM: C[M,Nc] = A[M,K](bf16) * W[Nc,K]^T(bf16) + bias
// Block 256 thr (8 waves) -> 128x64 tile. K-step 64, double-buffered LDS
// staged by async Global->LDS DMA. Compute: batch fragment loads, one
// dscnt wait, then an 8-deep back-to-back WMMA burst.
// ---------------------------------------------------------------------------
#define LDS_STRIDE 72   // 64 + 8 pad (bf16 elems) -> 144B rows, 16B aligned

__global__ __launch_bounds__(256, 2)
void gemm_bf16_wmma(const bf16* __restrict__ A, int lda,
                    const bf16* __restrict__ W, int ldw,
                    const float* __restrict__ bias,
                    bf16* __restrict__ outB, float* __restrict__ outF,
                    const float* __restrict__ resid,
                    int ldc, int K)
{
  __shared__ __align__(16) bf16 sA[2][128 * LDS_STRIDE];  // 2 x 18KB
  __shared__ __align__(16) bf16 sB[2][64  * LDS_STRIDE];  // 2 x  9KB

  const int tid  = threadIdx.x;
  const int lane = tid & 31;
  const int wave = tid >> 5;
  const int m0 = blockIdx.y * 128;
  const int n0 = blockIdx.x * 64;

  // fragment geometry (wave32; contiguous k = 16*(lane/16) + s, same in A and B)
  const int ar  = lane & 15;          // A: row within 16-row tile
  const int akb = (lane >> 4) * 16;   // A: k sub-chunk base
  const int bn  = lane & 15;          // B: column within 16-col tile
  const int bkb = (lane >> 4) * 16;   // B: k sub-chunk base

  auto issue_tile = [&](int buf, int k0) {
#pragma unroll
    for (int i = 0; i < 4; ++i) {                 // A: 1024 b128 chunks
      const int c = tid + i * 256;
      const int r = c >> 3, cc = (c & 7) * 8;
      cp16_g2l(&sA[buf][r * LDS_STRIDE + cc],
               A + (size_t)(m0 + r) * lda + k0 + cc);
    }
#pragma unroll
    for (int i = 0; i < 2; ++i) {                 // B: 512 b128 chunks
      const int c = tid + i * 256;
      const int r = c >> 3, cc = (c & 7) * 8;
      cp16_g2l(&sB[buf][r * LDS_STRIDE + cc],
               W + (size_t)(n0 + r) * ldw + k0 + cc);
    }
  };

  v8f acc[4];
#pragma unroll
  for (int t = 0; t < 4; ++t) acc[t] = zero8();

  issue_tile(0, 0);
  int buf = 0;
  for (int k0 = 0; k0 < K; k0 += 64) {
    const bool has_next = (k0 + 64) < K;
    if (has_next) issue_tile(buf ^ 1, k0 + 64);   // prefetch next tile
    // previous 6 copies (this tile) retire in order; next 6 stay in flight
    if (has_next) { WAIT_ASYNC(6); } else { WAIT_ASYNC(0); }
    __syncthreads();

    const bf16* aB = &sA[buf][(wave * 16 + ar) * LDS_STRIDE + akb];
    const bf16* wB = &sB[buf][bn * LDS_STRIDE + bkb];

    // batch all fragment loads, then burst the WMMAs
    v16bf af[2], bfr[2][4];
#pragma unroll
    for (int ks = 0; ks < 2; ++ks) {
      af[ks] = cat16(ld8(aB + ks * 32), ld8(aB + ks * 32 + 8));
#pragma unroll
      for (int t = 0; t < 4; ++t) {
        const bf16* wp = wB + t * 16 * LDS_STRIDE + ks * 32;
        bfr[ks][t] = cat16(ld8(wp), ld8(wp + 8));
      }
    }
#pragma unroll
    for (int ks = 0; ks < 2; ++ks)
#pragma unroll
      for (int t = 0; t < 4; ++t)
        acc[t] = wmma_bf16(af[ks], bfr[ks][t], acc[t]);

    __syncthreads();                              // tile fully consumed
    buf ^= 1;
  }

  // C layout: VGPR r, lane L -> M = r + (L>=16 ? 8 : 0), N = L&15
  const int cm = m0 + wave * 16 + (lane >> 4) * 8;
  const int cn = n0 + (lane & 15);
#pragma unroll
  for (int t = 0; t < 4; ++t) {
    const int n = cn + t * 16;
    const float bv = bias ? bias[n] : 0.f;
#pragma unroll
    for (int r = 0; r < 8; ++r) {
      float v = acc[t][r] + bv;
      const size_t idx = (size_t)(cm + r) * ldc + n;
      if (outF) {
        if (resid) v += resid[idx];
        outF[idx] = v;
      } else {
        outB[idx] = (bf16)v;
      }
    }
  }
}

// ---------------------------------------------------------------------------
// RoPE + de-interleave of fused QKV ([B,N,H,HD,3]) -> q/k/v [B,H,N,HD] bf16
// ---------------------------------------------------------------------------
__global__ __launch_bounds__(256)
void rope_scatter(const bf16* __restrict__ qkv, const float* __restrict__ enc,
                  bf16* __restrict__ qr, bf16* __restrict__ kr, bf16* __restrict__ vr)
{
  const int idx = blockIdx.x * blockDim.x + threadIdx.x;  // 2^22 total
  const int d2 = idx & 31;
  const int h  = (idx >> 5) & 15;
  const int n  = (idx >> 9) & 2047;
  const int b  = idx >> 20;

  const bf16* row = qkv + (size_t)(b * 2048 + n) * 3072 + h * 192 + d2 * 6;
  const float q0 = (float)row[0], k0 = (float)row[1], v0 = (float)row[2];
  const float q1 = (float)row[3], k1 = (float)row[4], v1 = (float)row[5];

  const size_t eb = ((size_t)b * 2048 + n) * 64 + d2 * 2;       // cos base
  const float c0 = enc[eb],          c1 = enc[eb + 1];
  const float s0 = enc[eb + 524288], s1 = enc[eb + 1 + 524288]; // sin base

  const size_t ob = ((size_t)((b * 16 + h) * 2048 + n)) * 64 + d2 * 2;
  qr[ob]     = (bf16)(q0 * c0 - q1 * s0);
  qr[ob + 1] = (bf16)(q1 * c1 + q0 * s1);
  kr[ob]     = (bf16)(k0 * c0 - k1 * s0);
  kr[ob + 1] = (bf16)(k1 * c1 + k0 * s1);
  vr[ob]     = (bf16)v0;
  vr[ob + 1] = (bf16)v1;
}

// ---------------------------------------------------------------------------
// Flash attention, HD=64, scale=1/8. 4 waves x 16 q-rows per block.
// K tile staged via async DMA; V transposed into LDS; P through LDS for
// C-layout -> A-layout. 8 WMMAs / 32 keys / wave, batched fragment loads.
// ---------------------------------------------------------------------------
__global__ __launch_bounds__(128, 2)
void flash_attn_wmma(const bf16* __restrict__ Q, const bf16* __restrict__ Km,
                     const bf16* __restrict__ V, bf16* __restrict__ ctx)
{
  __shared__ __align__(16) bf16 sK[32 * 64];     // [key][dim]
  __shared__ __align__(16) bf16 sVT[64 * 40];    // V^T [dim][key], pad 40
  __shared__ __align__(16) bf16 sP[4 * 16 * 32]; // per-wave P tiles

  const int bh = blockIdx.x;          // B*H = 64
  const int b  = bh >> 4, h = bh & 15;
  const size_t base = (size_t)bh * 2048 * 64;
  const int lane = threadIdx.x & 31, wave = threadIdx.x >> 5;
  const int q0 = blockIdx.y * 64 + wave * 16;

  const int ar = lane & 15, akb = (lane >> 4) * 16;
  const int bn = lane & 15, bkb = (lane >> 4) * 16;

  const bf16* qp = Q + base + (size_t)(q0 + ar) * 64 + akb;
  const v16bf qa0 = cat16(ld8(qp),      ld8(qp + 8));
  const v16bf qa1 = cat16(ld8(qp + 32), ld8(qp + 40));

  v8f o[4];
  float mrow[8], lrow[8];
#pragma unroll
  for (int f = 0; f < 4; ++f) o[f] = zero8();
#pragma unroll
  for (int r = 0; r < 8; ++r) { mrow[r] = -3.0e38f; lrow[r] = 0.f; }

  bf16* sPw = sP + wave * (16 * 32);
  const float scale = 0.125f;

  for (int kt = 0; kt < 2048; kt += 32) {
    __syncthreads();
    {
      const int t = threadIdx.x;
      const bf16* kg = Km + base + (size_t)kt * 64;
      cp16_g2l(&((bf16x8*)sK)[t],       &((const bf16x8*)kg)[t]);
      cp16_g2l(&((bf16x8*)sK)[t + 128], &((const bf16x8*)kg)[t + 128]);
      const int key = t >> 2, dimc = (t & 3) * 16;
      const bf16* vg = V + base + (size_t)(kt + key) * 64 + dimc;
      bf16x8 va = ld8(vg), vb = ld8(vg + 8);
#pragma unroll
      for (int i = 0; i < 8; ++i) {
        sVT[(dimc + i) * 40 + key]     = va[i];
        sVT[(dimc + 8 + i) * 40 + key] = vb[i];
      }
      WAIT_ASYNC(0);
    }
    __syncthreads();

    // S = Q K^T : batch all 4 K-fragments, then 4 WMMAs
    const bf16* kr0 = sK + bn * 64 + bkb;
    const bf16* kr1 = sK + (16 + bn) * 64 + bkb;
    v16bf kf[4];
    kf[0] = cat16(ld8(kr0),      ld8(kr0 + 8));
    kf[1] = cat16(ld8(kr0 + 32), ld8(kr0 + 40));
    kf[2] = cat16(ld8(kr1),      ld8(kr1 + 8));
    kf[3] = cat16(ld8(kr1 + 32), ld8(kr1 + 40));
    v8f s0 = zero8(), s1 = zero8();
    s0 = wmma_bf16(qa0, kf[0], s0);
    s1 = wmma_bf16(qa0, kf[2], s1);
    s0 = wmma_bf16(qa1, kf[1], s0);
    s1 = wmma_bf16(qa1, kf[3], s1);

    // issue V-fragment loads early; ds latency overlaps the softmax VALU work
    v16bf vf[4];
#pragma unroll
    for (int f = 0; f < 4; ++f) {
      const bf16* vp = sVT + (f * 16 + bn) * 40 + bkb;
      vf[f] = cat16(ld8(vp), ld8(vp + 8));
    }

    // online softmax: row-reduce within 16-lane half-wave groups
    float p0[8], p1[8];
#pragma unroll
    for (int r = 0; r < 8; ++r) {
      const float a = s0[r] * scale, bq = s1[r] * scale;
      float t = fmaxf(a, bq);
      t = fmaxf(t, __shfl_xor(t, 1));
      t = fmaxf(t, __shfl_xor(t, 2));
      t = fmaxf(t, __shfl_xor(t, 4));
      t = fmaxf(t, __shfl_xor(t, 8));
      const float mn = fmaxf(mrow[r], t);
      const float alpha = __expf(mrow[r] - mn);
      const float e0 = __expf(a - mn), e1 = __expf(bq - mn);
      float rs = e0 + e1;
      rs += __shfl_xor(rs, 1);
      rs += __shfl_xor(rs, 2);
      rs += __shfl_xor(rs, 4);
      rs += __shfl_xor(rs, 8);
      lrow[r] = lrow[r] * alpha + rs;
      mrow[r] = mn;
      p0[r] = e0; p1[r] = e1;
#pragma unroll
      for (int f = 0; f < 4; ++f) o[f][r] *= alpha;
    }

    // P: C-layout -> LDS (16x32 row-major) -> A-fragment
    const int prow = (lane >> 4) * 8;
#pragma unroll
    for (int r = 0; r < 8; ++r) {
      sPw[(prow + r) * 32 + bn]      = (bf16)p0[r];
      sPw[(prow + r) * 32 + 16 + bn] = (bf16)p1[r];
    }
    const bf16* pp = sPw + (lane & 15) * 32 + akb;
    v16bf pa = cat16(ld8(pp), ld8(pp + 8));

    // ctx += P V : 4-deep WMMA burst
#pragma unroll
    for (int f = 0; f < 4; ++f) o[f] = wmma_bf16(pa, vf[f], o[f]);
  }

  // normalize, store ctx as [B,N,H*64]
#pragma unroll
  for (int f = 0; f < 4; ++f) {
#pragma unroll
    for (int r = 0; r < 8; ++r) {
      const float v = o[f][r] / lrow[r];
      const int m = q0 + (lane >> 4) * 8 + r;
      const size_t oi = ((size_t)(b * 2048 + m)) * 1024 + h * 64 + f * 16 + bn;
      ctx[oi] = (bf16)v;
    }
  }
}

// ---------------------------------------------------------------------------
// LayerNorm (fp32 stats) + exact-erf GELU, bf16 in -> bf16 out. 1 row/block.
// ---------------------------------------------------------------------------
__global__ __launch_bounds__(256)
void ln_gelu_kernel(const bf16* __restrict__ hin, const float* __restrict__ gam,
                    const float* __restrict__ bet, bf16* __restrict__ out)
{
  const int row = blockIdx.x;
  const bf16* p = hin + (size_t)row * 2048;
  float s = 0.f, s2 = 0.f;
  for (int i = threadIdx.x; i < 2048; i += 256) {
    const float v = (float)p[i];
    s += v; s2 += v * v;
  }
#pragma unroll
  for (int off = 16; off >= 1; off >>= 1) {
    s  += __shfl_xor(s,  off);
    s2 += __shfl_xor(s2, off);
  }
  __shared__ float rs[8], rs2[8];
  const int wave = threadIdx.x >> 5, lane = threadIdx.x & 31;
  if (lane == 0) { rs[wave] = s; rs2[wave] = s2; }
  __syncthreads();
  float S = 0.f, S2 = 0.f;
#pragma unroll
  for (int w2 = 0; w2 < 8; ++w2) { S += rs[w2]; S2 += rs2[w2]; }
  const float mu   = S * (1.f / 2048.f);
  const float var  = S2 * (1.f / 2048.f) - mu * mu;
  const float rstd = rsqrtf(var + 1e-5f);
  for (int i = threadIdx.x; i < 2048; i += 256) {
    const float v = ((float)p[i] - mu) * rstd * gam[i] + bet[i];
    const float g = 0.5f * v * (1.f + erff(v * 0.70710678118654752f));
    out[(size_t)row * 2048 + i] = (bf16)g;
  }
}

// ---------------------------------------------------------------------------
// Launch
// ---------------------------------------------------------------------------
extern "C" void kernel_launch(void* const* d_in, const int* in_sizes, int n_in,
                              void* d_out, int out_size, void* d_ws, size_t ws_size,
                              hipStream_t stream)
{
  const float* x      = (const float*)d_in[0];
  const float* enc    = (const float*)d_in[1];
  const float* Wqkv_w = (const float*)d_in[2];
  const float* Wqkv_b = (const float*)d_in[3];
  const float* out_w  = (const float*)d_in[4];
  const float* out_b  = (const float*)d_in[5];
  const float* ffn1_w = (const float*)d_in[6];
  const float* ffn1_b = (const float*)d_in[7];
  const float* ln_g   = (const float*)d_in[8];
  const float* ln_b   = (const float*)d_in[9];
  const float* ffn2_w = (const float*)d_in[10];
  const float* ffn2_b = (const float*)d_in[11];

  char* w = (char*)d_ws;
  auto alloc = [&](size_t bytes) {
    char* p = w;
    w += (bytes + 255) & ~(size_t)255;
    return p;
  };
  bf16* xb   = (bf16*)alloc(8192ULL * 1024 * 2);
  bf16* wqkv = (bf16*)alloc(3072ULL * 1024 * 2);
  bf16* wout = (bf16*)alloc(1024ULL * 1024 * 2);
  bf16* wf1  = (bf16*)alloc(2048ULL * 2048 * 2);
  bf16* wf2  = (bf16*)alloc(1024ULL * 2048 * 2);
  bf16* qkvr = (bf16*)alloc(8192ULL * 3072 * 2);
  bf16* qr   = (bf16*)alloc(64ULL * 2048 * 64 * 2);
  bf16* kr   = (bf16*)alloc(64ULL * 2048 * 64 * 2);
  bf16* vr   = (bf16*)alloc(64ULL * 2048 * 64 * 2);
  bf16* ctxb = (bf16*)alloc(8192ULL * 1024 * 2);
  bf16* hb   = (bf16*)alloc(8192ULL * 2048 * 2);
  bf16* h1b  = qkvr;   // reuse: qkv dead after rope_scatter
  bf16* gb   = qr;     // reuse: q/k dead after flash (qr+kr = 33.5MB exactly)

  cvt_bf16<<<1024, 256, 0, stream>>>(x,      xb,   8192L * 1024);
  cvt_bf16<<<512,  256, 0, stream>>>(Wqkv_w, wqkv, 3072L * 1024);
  cvt_bf16<<<512,  256, 0, stream>>>(out_w,  wout, 1024L * 1024);
  cvt_bf16<<<512,  256, 0, stream>>>(ffn1_w, wf1,  2048L * 2048);
  cvt_bf16<<<512,  256, 0, stream>>>(ffn2_w, wf2,  1024L * 2048);
  cvt_bf16_strided<<<1024, 256, 0, stream>>>(x, hb, 8192, 1024, 2048);

  gemm_bf16_wmma<<<dim3(48, 64), 256, 0, stream>>>(
      xb, 1024, wqkv, 1024, Wqkv_b, qkvr, nullptr, nullptr, 3072, 1024);

  rope_scatter<<<(1 << 22) / 256, 256, 0, stream>>>(qkvr, enc, qr, kr, vr);

  flash_attn_wmma<<<dim3(64, 32), 128, 0, stream>>>(qr, kr, vr, ctxb);

  gemm_bf16_wmma<<<dim3(16, 64), 256, 0, stream>>>(
      ctxb, 1024, wout, 1024, out_b, hb + 1024, nullptr, nullptr, 2048, 1024);

  gemm_bf16_wmma<<<dim3(32, 64), 256, 0, stream>>>(
      hb, 2048, wf1, 2048, ffn1_b, h1b, nullptr, nullptr, 2048, 2048);

  ln_gelu_kernel<<<8192, 256, 0, stream>>>(h1b, ln_g, ln_b, gb);

  gemm_bf16_wmma<<<dim3(16, 64), 256, 0, stream>>>(
      gb, 2048, wf2, 2048, ffn2_b, nullptr, (float*)d_out, x, 1024, 2048);
}
